// ResidualQuantizer_55705725829368
// MI455X (gfx1250) — matrix operand order repaired and checked by reading, not compiled, not run on hardware
//
#include <hip/hip_runtime.h>

// ---------------- Problem constants (match reference) ----------------
#define BQ 16
#define TQ 4096
#define DQ 512
#define KQ 1024
#define QQ 8
#define ROWS (BQ * TQ)            // 65536 rows of dimension DQ
#define M_TILE 64                 // rows per workgroup
#define NUM_WG (ROWS / M_TILE)    // 1024 workgroups per stage
#define LDS_STRIDE (DQ + 8)       // padded bf16 row stride in LDS

typedef __bf16 bf16_t;
typedef __attribute__((ext_vector_type(16))) __bf16 v16bf;
typedef __attribute__((ext_vector_type(8)))  __bf16 v8bf;
typedef __attribute__((ext_vector_type(8)))  float  v8f;

union V16U { v16bf v; v8bf h[2]; };

// ---------------- Prep: codebooks fp32 -> bf16, and c2[k] = |c|^2 ----
__global__ __launch_bounds__(256) void rvq_prep(const float* __restrict__ codebooks,
                                                bf16_t* __restrict__ cb_bf16,
                                                float* __restrict__ c2) {
    __shared__ float red[256];
    const int row = blockIdx.x;                    // 0 .. Q*K-1
    const float* src = codebooks + (size_t)row * DQ;
    bf16_t* dst = cb_bf16 + (size_t)row * DQ;
    float s = 0.f;
    for (int c = threadIdx.x; c < DQ; c += 256) {
        float v = src[c];
        dst[c] = (bf16_t)v;
        s += v * v;
    }
    red[threadIdx.x] = s;
    __syncthreads();
    for (int o = 128; o >= 1; o >>= 1) {
        if ((int)threadIdx.x < o) red[threadIdx.x] += red[threadIdx.x + o];
        __syncthreads();
    }
    if (threadIdx.x == 0) c2[row] = red[0];
}

// ---------------- residual = x (float4 copy) -------------------------
__global__ void rvq_copy(const float4* __restrict__ src, float4* __restrict__ dst, int n4) {
    int i = blockIdx.x * blockDim.x + threadIdx.x;
    if (i < n4) dst[i] = src[i];
}

// ---------------- One quantizer stage --------------------------------
// dots via v_wmma_f32_16x16x32_bf16; per-row argmin over K=1024;
// residual[row] -= cb[argmin]; indices + per-block loss partials out.
__global__ __launch_bounds__(256) void rvq_stage(
    float* __restrict__ residual,                 // [ROWS][DQ] fp32 (in d_out)
    const bf16_t* __restrict__ cbq_bf16,          // [KQ][DQ] bf16
    const float* __restrict__ cbq_f32,            // [KQ][DQ] fp32
    const float* __restrict__ c2q,                // [KQ]
    float* __restrict__ outIdx,                   // [ROWS][QQ] (as float)
    float* __restrict__ lossPart,                 // [NUM_WG] (this stage's slice)
    int q)
{
    __shared__ __attribute__((aligned(32))) bf16_t sA[M_TILE * LDS_STRIDE];
    __shared__ float sR2[M_TILE];
    __shared__ float sRedV[8 * M_TILE];
    __shared__ int   sRedI[8 * M_TILE];
    __shared__ int   sMinIdx[M_TILE];
    __shared__ float sRowLoss[M_TILE];

    const int tid  = threadIdx.x;
    const int lane = tid & 31;
    const int wave = tid >> 5;
    const int col  = lane & 15;
    const int half = lane >> 4;
    const int row0 = blockIdx.x * M_TILE;

    // Phase 0: load fp32 residual tile, convert to bf16 in LDS, compute r2 per row.
    for (int r = 0; r < 8; ++r) {
        const int row = wave * 8 + r;
        const float* src = residual + (size_t)(row0 + row) * DQ + lane * 16;
        float vals[16];
        #pragma unroll
        for (int j = 0; j < 4; ++j) {
            float4 f = ((const float4*)src)[j];
            vals[4*j+0] = f.x; vals[4*j+1] = f.y; vals[4*j+2] = f.z; vals[4*j+3] = f.w;
        }
        float ss = 0.f;
        #pragma unroll
        for (int j = 0; j < 16; ++j) ss += vals[j] * vals[j];
        #pragma unroll
        for (int m = 16; m >= 1; m >>= 1) ss += __shfl_xor(ss, m, 32);
        if (lane == 0) sR2[row] = ss;
        __align__(16) bf16_t tmp[16];
        #pragma unroll
        for (int j = 0; j < 16; ++j) tmp[j] = (bf16_t)vals[j];
        v8bf* dst = (v8bf*)(sA + row * LDS_STRIDE + lane * 16);
        dst[0] = *(v8bf*)&tmp[0];
        dst[1] = *(v8bf*)&tmp[8];
    }
    __syncthreads();

    // Running per-(row) minima. rv[mb][i] tracks row mb*16 + (half?8:0) + i over
    // the columns this lane sees (n = tile*16 + col).
    float rv[4][8];
    int   ri[4][8];
    #pragma unroll
    for (int mb = 0; mb < 4; ++mb)
        #pragma unroll
        for (int i = 0; i < 8; ++i) { rv[mb][i] = 3.4e38f; ri[mb][i] = 0; }

    for (int pass = 0; pass < 4; ++pass) {
        const int nt0 = pass * 16 + wave * 2;     // this wave's 2 N-tiles
        v8f acc[4][2];
        #pragma unroll
        for (int mb = 0; mb < 4; ++mb)
            #pragma unroll
            for (int nt = 0; nt < 2; ++nt)
                acc[mb][nt] = (v8f){0.f,0.f,0.f,0.f,0.f,0.f,0.f,0.f};

        #pragma unroll 2
        for (int ks = 0; ks < 16; ++ks) {
            const int kb = ks * 32;
            // B operand (32x16 bf16): lane=column n; lanes 0-15 hold K kb..kb+15,
            // lanes 16-31 hold K kb+16..kb+31. Contiguous 32B per lane from row-major cb.
            const int koff = kb + half * 16;
            v16bf b0 = *(const v16bf*)(cbq_bf16 + (size_t)(nt0 * 16 + col) * DQ + koff);
            v16bf b1 = *(const v16bf*)(cbq_bf16 + (size_t)((nt0 + 1) * 16 + col) * DQ + koff);
            #pragma unroll
            for (int mb = 0; mb < 4; ++mb) {
                // A operand (16x32 bf16): lanes 0-15 K kb..kb+15,
                // lanes 16-31 K kb+16..kb+31, row m = mb*16 + col.
                const bf16_t* arow = sA + (mb * 16 + col) * LDS_STRIDE;
                V16U a;
                a.h[0] = *(const v8bf*)(arow + koff);
                a.h[1] = *(const v8bf*)(arow + koff + 8);
                acc[mb][0] = __builtin_amdgcn_wmma_f32_16x16x32_bf16(
                    false, a.v, false, b0, (short)0, acc[mb][0], false, false);
                acc[mb][1] = __builtin_amdgcn_wmma_f32_16x16x32_bf16(
                    false, a.v, false, b1, (short)0, acc[mb][1], false, false);
            }
        }

        // Fold scores: score = c2[n] - 2*dot  (r2 constant in n, irrelevant for argmin)
        #pragma unroll
        for (int nt = 0; nt < 2; ++nt) {
            const int n = (nt0 + nt) * 16 + col;
            const float c2v = c2q[n];
            #pragma unroll
            for (int mb = 0; mb < 4; ++mb)
                #pragma unroll
                for (int i = 0; i < 8; ++i) {
                    float s = fmaf(-2.f, acc[mb][nt][i], c2v);
                    if (s < rv[mb][i]) { rv[mb][i] = s; ri[mb][i] = n; }
                }
        }
    }

    // Cross-lane (within 16-lane half) min+argmin; ties -> smaller index.
    #pragma unroll
    for (int mb = 0; mb < 4; ++mb)
        #pragma unroll
        for (int i = 0; i < 8; ++i) {
            float v = rv[mb][i]; int ix = ri[mb][i];
            #pragma unroll
            for (int off = 8; off >= 1; off >>= 1) {
                float ov = __shfl_xor(v, off, 32);
                int   oi = __shfl_xor(ix, off, 32);
                if (ov < v || (ov == v && oi < ix)) { v = ov; ix = oi; }
            }
            rv[mb][i] = v; ri[mb][i] = ix;
        }
    if (col == 0) {
        const int rbase = half * 8;                // lane 0 -> rows m+0..7, lane 16 -> m+8..15
        #pragma unroll
        for (int mb = 0; mb < 4; ++mb)
            #pragma unroll
            for (int i = 0; i < 8; ++i) {
                const int r = mb * 16 + rbase + i;
                sRedV[wave * M_TILE + r] = rv[mb][i];
                sRedI[wave * M_TILE + r] = ri[mb][i];
            }
    }
    __syncthreads();

    // Cross-wave reduction (fixed order -> deterministic), write index + row loss.
    if (tid < M_TILE) {
        float bv = sRedV[tid]; int bi = sRedI[tid];
        #pragma unroll
        for (int w = 1; w < 8; ++w) {
            float v = sRedV[w * M_TILE + tid]; int ix = sRedI[w * M_TILE + tid];
            if (v < bv || (v == bv && ix < bi)) { bv = v; bi = ix; }
        }
        sMinIdx[tid] = bi;
        sRowLoss[tid] = sR2[tid] + bv;             // = min squared distance
        outIdx[(size_t)(row0 + tid) * QQ + q] = (float)bi;
    }
    __syncthreads();
    if (tid == 0) {
        float s = 0.f;
        for (int r = 0; r < M_TILE; ++r) s += sRowLoss[r];
        lossPart[blockIdx.x] = s;
    }

    // Residual update: residual[row] -= cb_f32[argmin]  (fp32, full precision)
    #pragma unroll
    for (int r = 0; r < 8; ++r) {
        const int row = wave * 8 + r;
        const int idx = sMinIdx[row];
        const float4* cb4  = (const float4*)(cbq_f32 + (size_t)idx * DQ + lane * 16);
        float4* res4 = (float4*)(residual + (size_t)(row0 + row) * DQ + lane * 16);
        #pragma unroll
        for (int j = 0; j < 4; ++j) {
            float4 a = res4[j]; float4 b = cb4[j];
            a.x -= b.x; a.y -= b.y; a.z -= b.z; a.w -= b.w;
            res4[j] = a;
        }
    }
}

// ---------------- Loss reduction: commit[q] = sum(partials)/ (B*T*D) -
__global__ __launch_bounds__(256) void rvq_loss(const float* __restrict__ lossPart,
                                                float* __restrict__ outLoss) {
    __shared__ float red[256];
    const int q = blockIdx.x;
    float s = 0.f;
    for (int i = threadIdx.x; i < NUM_WG; i += 256) s += lossPart[q * NUM_WG + i];
    red[threadIdx.x] = s;
    __syncthreads();
    for (int o = 128; o >= 1; o >>= 1) {
        if ((int)threadIdx.x < o) red[threadIdx.x] += red[threadIdx.x + o];
        __syncthreads();
    }
    if (threadIdx.x == 0)
        outLoss[q] = red[0] * (1.0f / ((float)BQ * (float)TQ * (float)DQ));  // COMMIT_W = 1
}

// ---------------- quantized_out = x - residual (in place in d_out) ----
__global__ void rvq_finalize(const float4* __restrict__ x4, float4* __restrict__ out4, int n4) {
    int i = blockIdx.x * blockDim.x + threadIdx.x;
    if (i < n4) {
        float4 a = x4[i], r = out4[i];
        out4[i] = make_float4(a.x - r.x, a.y - r.y, a.z - r.z, a.w - r.w);
    }
}

extern "C" void kernel_launch(void* const* d_in, const int* in_sizes, int n_in,
                              void* d_out, int out_size, void* d_ws, size_t ws_size,
                              hipStream_t stream) {
    const float* x         = (const float*)d_in[0];   // [B,T,D]
    const float* codebooks = (const float*)d_in[1];   // [Q,K,D]

    float* out      = (float*)d_out;
    float* residual = out;                                  // reuse quant section as residual
    float* outIdx   = out + (size_t)ROWS * DQ;              // [ROWS][Q] as float
    float* outLoss  = outIdx + (size_t)ROWS * QQ;           // [Q]

    // Workspace: bf16 codebooks (8 MB) + c2 (32 KB) + loss partials (32 KB)
    bf16_t* cb_bf16 = (bf16_t*)d_ws;
    float* c2       = (float*)(cb_bf16 + (size_t)QQ * KQ * DQ);
    float* lossPart = c2 + (size_t)QQ * KQ;

    const int n4 = ROWS * DQ / 4;                           // float4 count (divisible)

    rvq_copy<<<n4 / 256, 256, 0, stream>>>((const float4*)x, (float4*)residual, n4);
    rvq_prep<<<QQ * KQ, 256, 0, stream>>>(codebooks, cb_bf16, c2);

    for (int q = 0; q < QQ; ++q) {
        rvq_stage<<<NUM_WG, 256, 0, stream>>>(
            residual,
            cb_bf16 + (size_t)q * KQ * DQ,
            codebooks + (size_t)q * KQ * DQ,
            c2 + (size_t)q * KQ,
            outIdx,
            lossPart + (size_t)q * NUM_WG,
            q);
    }

    rvq_loss<<<QQ, 256, 0, stream>>>(lossPart, outLoss);
    rvq_finalize<<<n4 / 256, 256, 0, stream>>>((const float4*)x, (float4*)out, n4);
}
